// DeepGeoConvSNN_53429393162450
// MI455X (gfx1250) — matrix operand order repaired and checked by reading, compiled, not run on hardware
//
#include <hip/hip_runtime.h>

// ---------------------------------------------------------------------------
// Problem constants (B=128, C=64, T=480)
// ---------------------------------------------------------------------------
#define BB   128
#define TT   480
#define MTOT (BB * TT)   // 61440 rows, ordered m = t*128 + b

typedef __attribute__((ext_vector_type(2)))  float    v2f;
typedef __attribute__((ext_vector_type(8)))  float    v8f;
typedef __attribute__((ext_vector_type(16))) _Float16 v16h;

// ---------------------------------------------------------------------------
// K1: Mu = A_norm @ Wu^T, Mv = A_norm @ Wv^T   (two 64x64 matrices)
// ---------------------------------------------------------------------------
__global__ void mm_prep_kernel(const float* __restrict__ A,
                               const float* __restrict__ Wu,
                               const float* __restrict__ Wv,
                               float* __restrict__ Mu, float* __restrict__ Mv) {
  int gtid = blockIdx.x * blockDim.x + threadIdx.x;        // 8192 threads
  int which = gtid >> 12;
  int idx = gtid & 4095;
  int c = idx >> 6, o = idx & 63;
  const float* W = which ? Wv : Wu;
  float s = 0.f;
  for (int d = 0; d < 64; ++d) s += A[c * 64 + d] * W[o * 64 + d];
  (which ? Mv : Mu)[c * 64 + o] = s;
}

// ---------------------------------------------------------------------------
// K2: injection GEMM with V_WMMA_F32_16X16X4_F32 (exact fp32)
//   pre[m, o] = sum_c u[b,c,t]*Mu[c,o] + v[b,c,t]*Mv[c,o],  m = t*128+b
// Wave register tile: 16 rows x 64 cols (4 accumulators); A loaded once per
// k-step and reused across all 4 N-tiles and both (u,v) operands.
// ---------------------------------------------------------------------------
__global__ __launch_bounds__(128) void inj_gemm_kernel(
    const float* __restrict__ feat,
    const float* __restrict__ Mu, const float* __restrict__ Mv,
    float* __restrict__ pre) {
  int lane = threadIdx.x & 31;
  int m0 = blockIdx.x * 64 + (threadIdx.x >> 5) * 16;
  int row = lane & 15;
  int kh = lane >> 4;                 // K-half select (A/B f32 layout)
  int col = lane & 15;
  int rg = m0 + row;
  int t = rg >> 7;
  int b = rg & 127;
  const float* fb = feat + ((size_t)b * 64 * TT + t) * 4;  // +c*1920 per chan
  v8f acc[4] = {};
  for (int kk = 0; kk < 64; kk += 4) {
    int k0 = kk + 2 * kh;
    v2f au = { fb[(size_t)k0 * (TT * 4)],     fb[(size_t)(k0 + 1) * (TT * 4)] };
    v2f av = { fb[(size_t)k0 * (TT * 4) + 1], fb[(size_t)(k0 + 1) * (TT * 4) + 1] };
#pragma unroll
    for (int nt = 0; nt < 4; ++nt) {
      int n0 = nt * 16;
      v2f bu = { Mu[k0 * 64 + n0 + col], Mu[(k0 + 1) * 64 + n0 + col] };
      v2f bv = { Mv[k0 * 64 + n0 + col], Mv[(k0 + 1) * 64 + n0 + col] };
      acc[nt] = __builtin_amdgcn_wmma_f32_16x16x4_f32(false, au, false, bu,
                                                      (short)0, acc[nt], false, false);
      acc[nt] = __builtin_amdgcn_wmma_f32_16x16x4_f32(false, av, false, bv,
                                                      (short)0, acc[nt], false, false);
    }
  }
  int rbase = m0 + 8 * kh;            // C/D layout: VGPR r -> M = r + 8*half
#pragma unroll
  for (int nt = 0; nt < 4; ++nt)
#pragma unroll
    for (int r = 0; r < 8; ++r)
      pre[(size_t)(rbase + r) * 64 + nt * 16 + col] = acc[nt][r];
}

// ---------------------------------------------------------------------------
// K3: per-timestep BatchNorm over batch (mean/var over B for each (o,t)),
// normalizes pre in place. One block per t.
// ---------------------------------------------------------------------------
__global__ __launch_bounds__(256) void bn_inj_kernel(
    float* __restrict__ pre,
    const float* __restrict__ g, const float* __restrict__ be) {
  int t = blockIdx.x;
  int tid = threadIdx.x;
  int o = tid & 63, bg = tid >> 6;                     // 4 batch-groups
  __shared__ float ls[256], lss[256], mS[64], rS[64];
  float s = 0.f, ss = 0.f;
  for (int b = bg * 32; b < bg * 32 + 32; ++b) {
    float x = pre[((size_t)t * BB + b) * 64 + o];
    s += x; ss += x * x;
  }
  ls[tid] = s; lss[tid] = ss;
  __syncthreads();
  if (bg == 0) {
    for (int gg = 1; gg < 4; ++gg) { s += ls[o + gg * 64]; ss += lss[o + gg * 64]; }
    float m = s * (1.f / BB);
    float var = ss * (1.f / BB) - m * m;
    mS[o] = m; rS[o] = rsqrtf(var + 1e-5f);
  }
  __syncthreads();
  float m = mS[o], r = rS[o], ga = g[o], bb = be[o];
  for (int b = bg * 32; b < bg * 32 + 32; ++b) {
    size_t idx = ((size_t)t * BB + b) * 64 + o;
    pre[idx] = (pre[idx] - m) * r * ga + bb;
  }
}

// ---------------------------------------------------------------------------
// K4: LIF scan 1. Thread per (b,c); decay computed on the fly from curv/tang.
// spikes written as f16 (exact: {0,1}) in [T,B,64].
// ---------------------------------------------------------------------------
__global__ __launch_bounds__(256) void lif1_kernel(
    const float* __restrict__ inj, const float* __restrict__ feat,
    _Float16* __restrict__ sp1) {
  int gtid = blockIdx.x * blockDim.x + threadIdx.x;     // 8192 threads
  int c = gtid & 63, b = gtid >> 6;
  float v = 0.f;
  const float* fb = feat + ((size_t)(b * 64 + c) * TT) * 4;
  for (int t = 0; t < TT; ++t) {
    float i = inj[((size_t)t * BB + b) * 64 + c];
    float cu = fb[t * 4 + 2], ta = fb[t * 4 + 3];
    float d = __expf(-__expf(0.8f * cu + 0.4f * ta) * (1.f / 35.f));
    v = v * d + i;
    float s = (v > 0.15f) ? 1.f : 0.f;
    v -= s * 0.15f;
    sp1[((size_t)t * BB + b) * 64 + c] = (_Float16)s;
  }
}

// ---------------------------------------------------------------------------
// K5: weight prep -> WMMA-B-operand swizzled f16 tiles.
// B operand layout (32x16 f16): lane col=lane&15, Kbase=(lane>>4)*16,
// per-lane halfs j=0..15 hold W[Kbase+j, col]. One v16h per lane per tile.
// Tile index = (k*(Cin/32) + kcb)*(Cout/16) + ntile ; 512 halfs per tile.
// ---------------------------------------------------------------------------
__global__ void prep_w_kernel(const float* __restrict__ w,
                              _Float16* __restrict__ wp,
                              int Cout, int Cin, int taps, int total) {
  int tid = blockIdx.x * blockDim.x + threadIdx.x;
  if (tid >= total) return;
  int ntiles = Cout >> 4, kcbn = Cin >> 5;
  int lane = tid & 31, tile = tid >> 5;
  int ntile = tile % ntiles;
  int rest = tile / ntiles;
  int kcb = rest % kcbn, k = rest / kcbn;
  int col = lane & 15, Kb = (lane >> 4) * 16, kc = kcb << 5;
  int o = ntile * 16 + col;
  _Float16* dst = wp + (size_t)tile * 512 + lane * 16;
  for (int j = 0; j < 16; ++j) {
    int c = kc + Kb + j;
    dst[j] = (_Float16)w[((size_t)o * Cin + c) * taps + k];
  }
}

// ---------------------------------------------------------------------------
// K6: temporal conv as TAPS accumulated GEMMs on V_WMMA_F32_16X16X32_F16,
// FUSED with the 1x1 shortcut conv (fires at the center tap, its own accs).
// sp: [T, B, CIN] f16 spikes (exact). outc/outs: [T*B, COUT] f32.
// Wave register tile: 16 rows x 64 cols (4 accumulators each for conv + sc);
// one A load feeds 4 (or 8) WMMAs. Tap predication is wave-uniform.
// ---------------------------------------------------------------------------
template <int CIN, int COUT, int TAPS, int PAD>
__global__ __launch_bounds__(128) void spike_conv_fused_kernel(
    const _Float16* __restrict__ sp, const _Float16* __restrict__ wp,
    const _Float16* __restrict__ scp, float* __restrict__ outc,
    float* __restrict__ outs) {
  constexpr int KCBN = CIN / 32;
  constexpr int NTILES = COUT / 16;
  int lane = threadIdx.x & 31;
  int m0 = blockIdx.x * 64 + (threadIdx.x >> 5) * 16;
  int ng0 = blockIdx.y * 4;                 // first of 4 n-tiles for this wave
  int t = m0 >> 7;
  int b = (m0 & 127) + (lane & 15);
  int abase = (lane >> 4) * 16;             // A f16 K-half base
  v8f acc[4] = {};
  v8f asc[4] = {};
#pragma unroll
  for (int k = 0; k < TAPS; ++k) {
    int tt = t + k - PAD;
    if ((unsigned)tt >= (unsigned)TT) continue;     // wave-uniform
    const _Float16* sr = sp + (size_t)(tt * BB + b) * CIN;
    if (k + 1 < TAPS)
      __builtin_prefetch(sr + (size_t)BB * CIN, 0, 1);   // next tap's row
#pragma unroll
    for (int kcb = 0; kcb < KCBN; ++kcb) {
      int kc = kcb << 5;
      union { v16h v; float4 f[2]; } Am;
      // A 16x32 f16: lane row=lane&15; halfs 0..15 -> K=abase+j
      //              (two contiguous 16B loads)
      Am.f[0] = *(const float4*)(sr + kc + abase);
      Am.f[1] = *(const float4*)(sr + kc + abase + 8);
#pragma unroll
      for (int nt = 0; nt < 4; ++nt) {
        union { v16h v; float4 f[2]; } Bm;
        const _Float16* wt =
            wp + ((size_t)((k * KCBN + kcb) * NTILES + ng0 + nt) * 512) +
            lane * 16;
        Bm.f[0] = *(const float4*)(wt);
        Bm.f[1] = *(const float4*)(wt + 8);
        acc[nt] = __builtin_amdgcn_wmma_f32_16x16x32_f16(
            false, Am.v, false, Bm.v, (short)0, acc[nt], false, false);
      }
      if (k == PAD) {                      // shortcut 1x1 conv at center tap
#pragma unroll
        for (int nt = 0; nt < 4; ++nt) {
          union { v16h v; float4 f[2]; } Bs;
          const _Float16* st =
              scp + ((size_t)(kcb * NTILES + ng0 + nt) * 512) + lane * 16;
          Bs.f[0] = *(const float4*)(st);
          Bs.f[1] = *(const float4*)(st + 8);
          asc[nt] = __builtin_amdgcn_wmma_f32_16x16x32_f16(
              false, Am.v, false, Bs.v, (short)0, asc[nt], false, false);
        }
      }
    }
  }
  int colb = lane & 15;
  int rbase = m0 + 8 * (lane >> 4);
#pragma unroll
  for (int nt = 0; nt < 4; ++nt) {
    int col = (ng0 + nt) * 16 + colb;
#pragma unroll
    for (int r = 0; r < 8; ++r) {
      outc[(size_t)(rbase + r) * COUT + col] = acc[nt][r];
      outs[(size_t)(rbase + r) * COUT + col] = asc[nt][r];
    }
  }
}

// ---------------------------------------------------------------------------
// K7a/K7b: deterministic per-channel sum/sumsq over all (b,t) rows.
// ---------------------------------------------------------------------------
__global__ __launch_bounds__(256) void chan_partial_kernel(
    const float* __restrict__ X, float* __restrict__ part, int Cch) {
  int tid = threadIdx.x;
  int c = tid % Cch;
  int rg = tid / Cch;
  int nrg = 256 / Cch;
  int rows = MTOT / gridDim.x;                          // 240 blocks -> 256
  int r0 = blockIdx.x * rows;
  float s = 0.f, ss = 0.f;
  for (int r = r0 + rg; r < r0 + rows; r += nrg) {
    float x = X[(size_t)r * Cch + c];
    s += x; ss += x * x;
  }
  __shared__ float ls[256], lss[256];
  ls[tid] = s; lss[tid] = ss;
  __syncthreads();
  if (rg == 0) {
    for (int g = 1; g < nrg; ++g) { s += ls[c + g * Cch]; ss += lss[c + g * Cch]; }
    part[(size_t)blockIdx.x * Cch + c] = s;
    part[(size_t)240 * Cch + (size_t)blockIdx.x * Cch + c] = ss;
  }
}

__global__ void chan_final_kernel(const float* __restrict__ part,
                                  float* __restrict__ red, int Cch) {
  int c = threadIdx.x;
  if (c >= Cch) return;
  float s = 0.f, ss = 0.f;
  for (int blk = 0; blk < 240; ++blk) {
    s  += part[(size_t)blk * Cch + c];
    ss += part[(size_t)240 * Cch + (size_t)blk * Cch + c];
  }
  red[c] = s;
  red[Cch + c] = ss;
}

// ---------------------------------------------------------------------------
// K8: LIF scan 2 (decay=0.5, vth=0.3) with BN + shortcut folded in.
// ---------------------------------------------------------------------------
__global__ __launch_bounds__(256) void lif2_kernel(
    const float* __restrict__ c1, const float* __restrict__ s1,
    const float* __restrict__ red, const float* __restrict__ g,
    const float* __restrict__ be, const float* __restrict__ scb,
    _Float16* __restrict__ sp2) {
  int gtid = blockIdx.x * blockDim.x + threadIdx.x;     // 16384 threads
  int o = gtid & 127, b = gtid >> 7;
  float m = red[o] * (1.f / MTOT);
  float var = red[128 + o] * (1.f / MTOT) - m * m;
  float rstd = rsqrtf(var + 1e-5f);
  float ga = g[o], bb = be[o], sb = scb[o];
  float v = 0.f;
  for (int t = 0; t < TT; ++t) {
    size_t idx = ((size_t)t * BB + b) * 128 + o;
    float i = (c1[idx] - m) * rstd * ga + bb + s1[idx] + sb;
    v = v * 0.5f + i;
    float s = (v > 0.3f) ? 1.f : 0.f;
    v -= s * 0.3f;
    sp2[idx] = (_Float16)s;
  }
}

// ---------------------------------------------------------------------------
// K9: LIF scan 3 fused with AvgPool1d(32,32); writes flat[b, o*15+tb].
// ---------------------------------------------------------------------------
__global__ __launch_bounds__(256) void lif3_pool_kernel(
    const float* __restrict__ c2, const float* __restrict__ s2,
    const float* __restrict__ red, const float* __restrict__ g,
    const float* __restrict__ be, const float* __restrict__ scb,
    float* __restrict__ flat) {
  int gtid = blockIdx.x * blockDim.x + threadIdx.x;     // 32768 threads
  int o = gtid & 255, b = gtid >> 8;
  float m = red[o] * (1.f / MTOT);
  float var = red[256 + o] * (1.f / MTOT) - m * m;
  float rstd = rsqrtf(var + 1e-5f);
  float ga = g[o], bb = be[o], sb = scb[o];
  float v = 0.f;
  for (int tb = 0; tb < 15; ++tb) {
    float acc = 0.f;
    for (int gg = 0; gg < 32; ++gg) {
      int t = tb * 32 + gg;
      size_t idx = ((size_t)t * BB + b) * 256 + o;
      float i = (c2[idx] - m) * rstd * ga + bb + s2[idx] + sb;
      v = v * 0.5f + i;
      float s = (v > 0.3f) ? 1.f : 0.f;
      v -= s * 0.3f;
      acc += s;
    }
    flat[(size_t)b * 3840 + o * 15 + tb] = acc * (1.f / 32.f);
  }
}

// ---------------------------------------------------------------------------
// K10: BatchNorm over batch per flat feature (3840 features, 128 batch).
// ---------------------------------------------------------------------------
__global__ __launch_bounds__(128) void bn_flat_kernel(
    float* __restrict__ flat, const float* __restrict__ g,
    const float* __restrict__ be) {
  int f = blockIdx.x;
  int tid = threadIdx.x;
  __shared__ float ss[128], s2[128];
  float x = flat[(size_t)tid * 3840 + f];
  ss[tid] = x; s2[tid] = x * x;
  __syncthreads();
  for (int s = 64; s > 0; s >>= 1) {
    if (tid < s) { ss[tid] += ss[tid + s]; s2[tid] += s2[tid + s]; }
    __syncthreads();
  }
  float m = ss[0] * (1.f / 128.f);
  float var = s2[0] * (1.f / 128.f) - m * m;
  float rstd = rsqrtf(var + 1e-5f);
  flat[(size_t)tid * 3840 + f] = (x - m) * rstd * g[f] + be[f];
}

// ---------------------------------------------------------------------------
// K11: final FC [128,3840] x [4,3840]^T + fc_b -> out [128,4]
// ---------------------------------------------------------------------------
__global__ __launch_bounds__(256) void fc_kernel(
    const float* __restrict__ flat, const float* __restrict__ w,
    const float* __restrict__ bias, float* __restrict__ out) {
  int b = blockIdx.x >> 2, j = blockIdx.x & 3;
  int tid = threadIdx.x;
  float a = 0.f;
  for (int f = tid; f < 3840; f += 256)
    a += flat[(size_t)b * 3840 + f] * w[(size_t)j * 3840 + f];
  __shared__ float red[256];
  red[tid] = a;
  __syncthreads();
  for (int s = 128; s > 0; s >>= 1) {
    if (tid < s) red[tid] += red[tid + s];
    __syncthreads();
  }
  if (tid == 0) out[b * 4 + j] = red[0] + bias[j];
}

// ---------------------------------------------------------------------------
extern "C" void kernel_launch(void* const* d_in, const int* in_sizes, int n_in,
                              void* d_out, int out_size, void* d_ws, size_t ws_size,
                              hipStream_t stream) {
  const float* feat = (const float*)d_in[0];
  const float* A    = (const float*)d_in[1];
  const float* Wu_w = (const float*)d_in[2];
  const float* Wv_w = (const float*)d_in[4];
  const float* bng  = (const float*)d_in[6];
  const float* bnb  = (const float*)d_in[7];
  const float* c1w  = (const float*)d_in[8];
  const float* bn1g = (const float*)d_in[10];
  const float* bn1b = (const float*)d_in[11];
  const float* sc1w = (const float*)d_in[12];
  const float* sc1b = (const float*)d_in[13];
  const float* c2w  = (const float*)d_in[14];
  const float* bn2g = (const float*)d_in[16];
  const float* bn2b = (const float*)d_in[17];
  const float* sc2w = (const float*)d_in[18];
  const float* sc2b = (const float*)d_in[19];
  const float* pfg  = (const float*)d_in[20];
  const float* pfb  = (const float*)d_in[21];
  const float* fcw  = (const float*)d_in[22];
  const float* fcb  = (const float*)d_in[23];
  float* out = (float*)d_out;

  char* base = (char*)d_ws;
  size_t off = 0;
  auto alloc = [&](size_t n) -> void* {
    void* p = base + off;
    off = (off + n + 255) & ~(size_t)255;
    return p;
  };
  float*    Mu   = (float*)alloc(64 * 64 * 4);
  float*    Mv   = (float*)alloc(64 * 64 * 4);
  float*    red1 = (float*)alloc(256 * 4);
  float*    red2 = (float*)alloc(512 * 4);
  float*    part = (float*)alloc((size_t)240 * 256 * 2 * 4);
  _Float16* w1p  = (_Float16*)alloc((size_t)240 * 512 * 2);   // 15*2*8 tiles
  _Float16* s1p  = (_Float16*)alloc((size_t)16  * 512 * 2);   // 1*2*8
  _Float16* w2p  = (_Float16*)alloc((size_t)448 * 512 * 2);   // 7*4*16
  _Float16* s2p  = (_Float16*)alloc((size_t)64  * 512 * 2);   // 1*4*16
  float*    pre  = (float*)alloc((size_t)MTOT * 64 * 4);
  _Float16* sp1  = (_Float16*)alloc((size_t)MTOT * 64 * 2);
  float*    c1   = (float*)alloc((size_t)MTOT * 128 * 4);
  float*    s1   = (float*)alloc((size_t)MTOT * 128 * 4);
  _Float16* sp2  = (_Float16*)alloc((size_t)MTOT * 128 * 2);
  float*    c2   = (float*)alloc((size_t)MTOT * 256 * 4);
  float*    s2   = (float*)alloc((size_t)MTOT * 256 * 4);
  float*    flat = (float*)alloc((size_t)128 * 3840 * 4);

  // stage 0: folded matrices + weight swizzles (independent)
  mm_prep_kernel<<<32, 256, 0, stream>>>(A, Wu_w, Wv_w, Mu, Mv);
  prep_w_kernel<<<30, 256, 0, stream>>>(c1w,  w1p, 128, 64, 15, 7680);
  prep_w_kernel<<<2,  256, 0, stream>>>(sc1w, s1p, 128, 64, 1,  512);
  prep_w_kernel<<<56, 256, 0, stream>>>(c2w,  w2p, 256, 128, 7, 14336);
  prep_w_kernel<<<8,  256, 0, stream>>>(sc2w, s2p, 256, 128, 1, 2048);

  // stage 1: injection GEMM (f32 WMMA), per-t BN, LIF scan 1
  inj_gemm_kernel<<<MTOT / 64, 128, 0, stream>>>(feat, Mu, Mv, pre);
  bn_inj_kernel<<<TT, 256, 0, stream>>>(pre, bng, bnb);
  lif1_kernel<<<32, 256, 0, stream>>>(pre, feat, sp1);

  // stage 2: conv1 (K=15) + fused shortcut (f16 WMMA), BN stats, LIF scan 2
  spike_conv_fused_kernel<64, 128, 15, 7>
      <<<dim3(MTOT / 64, 2), 128, 0, stream>>>(sp1, w1p, s1p, c1, s1);
  chan_partial_kernel<<<240, 256, 0, stream>>>(c1, part, 128);
  chan_final_kernel<<<1, 256, 0, stream>>>(part, red1, 128);
  lif2_kernel<<<64, 256, 0, stream>>>(c1, s1, red1, bn1g, bn1b, sc1b, sp2);

  // stage 3: conv2 (K=7) + fused shortcut, BN stats, LIF scan 3 + pooling
  spike_conv_fused_kernel<128, 256, 7, 3>
      <<<dim3(MTOT / 64, 4), 128, 0, stream>>>(sp2, w2p, s2p, c2, s2);
  chan_partial_kernel<<<240, 256, 0, stream>>>(c2, part, 256);
  chan_final_kernel<<<1, 256, 0, stream>>>(part, red2, 256);
  lif3_pool_kernel<<<128, 256, 0, stream>>>(c2, s2, red2, bn2g, bn2b, sc2b, flat);

  // head: flat BN + FC
  bn_flat_kernel<<<3840, 128, 0, stream>>>(flat, pfg, pfb);
  fc_kernel<<<512, 256, 0, stream>>>(flat, fcw, fcb, out);
}
